// MultiHeadAttention_27891517620625
// MI455X (gfx1250) — matrix operand order repaired and checked by reading, not compiled, not run on hardware
//
#include <hip/hip_runtime.h>
#include <hip/hip_bf16.h>

typedef __attribute__((ext_vector_type(16))) _Float16 v16h;
typedef __attribute__((ext_vector_type(8)))  _Float16 v8h;
typedef __attribute__((ext_vector_type(8)))  float    v8f;
typedef __attribute__((ext_vector_type(4)))  float    v4f;
typedef __attribute__((ext_vector_type(4)))  unsigned int u32x4;
typedef __attribute__((ext_vector_type(8)))  int      i32x8;
typedef __attribute__((ext_vector_type(4)))  int      i32x4;

#define NB   32      // batch
#define SS   512     // sequence
#define DMODEL 128
#define NHEAD  8
#define DHEAD  64
#define HD   512     // NHEAD*DHEAD
#define NEGV (-1e9f)
#define EPSV (1e-5f)
#define SCALE 0.125f // 1/sqrt(64)

// ---------------- WMMA fragment loaders (CDNA5 16x16x32 f16, wave32) ----------
// A (16xK=32, M-row per lane): lane L holds row L%16; halves K = kb..kb+7 and
// kb+16..kb+23 where kb = 8*(L/16).  Two 16B contiguous loads.
static __device__ __forceinline__ v16h frag_a(const _Float16* base, int ld, int lane) {
  const _Float16* p = base + (lane & 15) * ld + ((lane >> 4) << 3);
  v8h lo = *(const v8h*)(p);
  v8h hi = *(const v8h*)(p + 16);
  v16h r;
#pragma unroll
  for (int i = 0; i < 8; ++i) { r[i] = lo[i]; r[i + 8] = hi[i]; }
  return r;
}
// B (K=32 x N=16) staged N-major (base[n*ld + k]): lane L holds column L%16,
// halves K = kb..kb+7 and kb+16..kb+23 where kb = 8*(L/16).  Two 16B contiguous loads.
static __device__ __forceinline__ v16h frag_b(const _Float16* base, int ld, int lane) {
  const _Float16* p = base + (lane & 15) * ld + ((lane >> 4) << 3);
  v8h lo = *(const v8h*)(p);
  v8h hi = *(const v8h*)(p + 16);
  v16h r;
#pragma unroll
  for (int i = 0; i < 8; ++i) { r[i] = lo[i]; r[i + 8] = hi[i]; }
  return r;
}
static __device__ __forceinline__ v8f wmma_f16(v16h a, v16h b, v8f c) {
  return __builtin_amdgcn_wmma_f32_16x16x32_f16(false, a, false, b, (short)0, c, false, false);
}

// Low 32 bits of a generic shared pointer == wave-relative LDS byte address
static __device__ __forceinline__ unsigned lds_addr32(const void* p) {
  return (unsigned)(unsigned long long)p;
}

// ---- Async global->LDS copy of 16 bytes (GLOBAL_LOAD_ASYNC_TO_LDS_B128) ------
static __device__ __forceinline__ void async_copy_b128(unsigned lds_off, const void* gptr) {
  asm volatile("global_load_async_to_lds_b128 %0, %1, off"
               :: "v"(lds_off), "v"(gptr) : "memory");
}
static __device__ __forceinline__ void wait_async0() {
  asm volatile("s_wait_asynccnt 0" ::: "memory");
}

// ---- TDM: contiguous 1-row tile load (data_size=2B), 6-arg builtin form ------
static __device__ __forceinline__ void tdm_load_linear(const void* gsrc, unsigned lds_off,
                                                       unsigned nhalves) {
#if __has_builtin(__builtin_amdgcn_tensor_load_to_lds)
  unsigned long long ga = (unsigned long long)gsrc;
  u32x4 g0;
  g0[0] = 1u;                                       // count=1 (valid descriptor)
  g0[1] = lds_off;                                  // D#.lds_addr
  g0[2] = (unsigned)ga;                             // D#.global_addr[31:0]
  g0[3] = (unsigned)((ga >> 32) & 0x1FFFFFFu) | (2u << 30); // addr[56:32] | type=2
  i32x8 g1;
  g1[0] = (int)(1u << 16);                          // data_size=1 (2 bytes)
  g1[1] = (int)(nhalves << 16);                     // tensor_dim0[15:0]
  g1[2] = (int)(1u << 16);                          // tensor_dim1=1
  g1[3] = (int)(nhalves << 16);                     // tile_dim0
  g1[4] = 1;                                        // tile_dim1=1
  g1[5] = (int)nhalves;                             // tensor_dim0_stride
  g1[6] = 0;
  g1[7] = 0;
  i32x4 gz4 = {0, 0, 0, 0};
  i32x8 gz8 = {0, 0, 0, 0, 0, 0, 0, 0};
  __builtin_amdgcn_tensor_load_to_lds(g0, g1, gz4, gz4, gz8, 0);
#else
  (void)gsrc; (void)lds_off; (void)nhalves;
#endif
}
static __device__ __forceinline__ void wait_tensor0() {
#if __has_builtin(__builtin_amdgcn_s_wait_tensorcnt)
  __builtin_amdgcn_s_wait_tensorcnt(0);
#else
  asm volatile("s_wait_tensorcnt 0" ::: "memory");
#endif
}

// ---------------- Kernel 0: weight convert+transpose to f16 N-major ----------
__global__ __launch_bounds__(256) void prep_weights_kernel(
    const float* __restrict__ Wq, const float* __restrict__ Wk,
    const float* __restrict__ Wv, const float* __restrict__ Wfc,
    _Float16* __restrict__ Wtq, _Float16* __restrict__ Wtk,
    _Float16* __restrict__ Wtv, _Float16* __restrict__ Wfct) {
  int i = blockIdx.x * 256 + threadIdx.x;          // 65536 threads
  int n  = i >> 7,  k  = i & 127;                  // QKV: [512][128] <- [128][512]
  Wtq[i] = (_Float16)Wq[k * HD + n];
  Wtk[i] = (_Float16)Wk[k * HD + n];
  Wtv[i] = (_Float16)Wv[k * HD + n];
  int nf = i >> 9,  kf = i & 511;                  // fc: [128][512] <- [512][128]
  Wfct[i] = (_Float16)Wfc[kf * DMODEL + nf];
}

// ---------------- Kernel 1: fused QKV projection ------------------------------
__global__ __launch_bounds__(256) void qkv_proj_kernel(
    const float* __restrict__ inQ, const float* __restrict__ inK,
    const float* __restrict__ inV,
    const _Float16* __restrict__ Wtq, const _Float16* __restrict__ Wtk,
    const _Float16* __restrict__ Wtv,
    _Float16* __restrict__ Qh, _Float16* __restrict__ Kh,
    _Float16* __restrict__ Vt) {
  __shared__ __align__(16) _Float16 Ain[3][16][DMODEL];   // 12 KB
  int bid = blockIdx.x;                 // NB * (SS/16) = 1024
  int b = bid >> 5;
  int s0 = (bid & 31) << 4;
  int tid = threadIdx.x, lane = tid & 31, w = tid >> 5;

  // stage 16x128 f32 rows of each input as f16
#pragma unroll
  for (int m = 0; m < 3; ++m) {
    const float* src = (m == 0 ? inQ : m == 1 ? inK : inV) + (b * SS + s0) * DMODEL;
#pragma unroll
    for (int i = 0; i < 2; ++i) {
      int idx = tid + i * 256;                       // 512 float4 per tile
      v4f v = ((const v4f*)src)[idx];
      _Float16* dst = &Ain[m][0][0] + idx * 4;
      dst[0] = (_Float16)v[0]; dst[1] = (_Float16)v[1];
      dst[2] = (_Float16)v[2]; dst[3] = (_Float16)v[3];
    }
  }
  __syncthreads();

#pragma unroll
  for (int m = 0; m < 3; ++m) {
    const _Float16* Wt = (m == 0 ? Wtq : m == 1 ? Wtk : Wtv);
    for (int nt = w; nt < 32; nt += 8) {             // 32 n-tiles of 16 cols
      v8f acc = {};
#pragma unroll
      for (int kc = 0; kc < 4; ++kc) {               // K = 128
        v16h a  = frag_a(&Ain[m][0][0] + kc * 32, DMODEL, lane);
        v16h bf = frag_b(Wt + nt * 16 * DMODEL + kc * 32, DMODEL, lane);
        acc = wmma_f16(a, bf, acc);
      }
      int colg = (nt << 4) + (lane & 15);
      int h = colg >> 6, d = colg & 63;
      int rbase = (lane >> 4) << 3;
#pragma unroll
      for (int r = 0; r < 8; ++r) {
        int s = s0 + rbase + r;
        _Float16 v = (_Float16)acc[r];
        if (m == 0)      Qh[((b * NHEAD + h) * SS + s) * DHEAD + d] = v;
        else if (m == 1) Kh[((b * NHEAD + h) * SS + s) * DHEAD + d] = v;
        else             Vt[((b * NHEAD + h) * DHEAD + d) * SS + s] = v;
      }
    }
  }
}

// ---------------- Kernel 2: fused scores + softmax + context ------------------
__global__ __launch_bounds__(128) void attn_kernel(
    const _Float16* __restrict__ Qh, const _Float16* __restrict__ Kh,
    const _Float16* __restrict__ Vt, const unsigned char* __restrict__ mask,
    float* __restrict__ attn_out, _Float16* __restrict__ ctx) {
  __shared__ __align__(16) _Float16 Qt[16][DHEAD];     // 2 KB
  __shared__ __align__(16) float    sc[16][SS];        // 32 KB
  __shared__ __align__(16) _Float16 P16[16][SS];       // 16 KB
  __shared__ float pmax[16][8], psum[16][8];

  int bid = blockIdx.x;                  // NB*NHEAD*(SS/16) = 8192
  int qt = bid & 31, h = (bid >> 5) & 7, b = bid >> 8;
  int q0 = qt << 4;
  int tid = threadIdx.x, lane = tid & 31, w = tid >> 5;

  // stage Q tile (16x64 f16 = 2 KB contiguous) via Tensor Data Mover
  const _Float16* Qbase = Qh + ((b * NHEAD + h) * SS + q0) * DHEAD;
#if __has_builtin(__builtin_amdgcn_tensor_load_to_lds)
  if (w == 0) {
    tdm_load_linear(Qbase, lds_addr32(&Qt[0][0]), 16 * DHEAD);
    wait_tensor0();
  }
#else
  ((uint4*)&Qt[0][0])[tid] = ((const uint4*)Qbase)[tid];
#endif
  __syncthreads();

  // ---- scores: S_qk = Q K^T * scale, masked -> LDS f32 ----
  const _Float16* Kbase = Kh + (b * NHEAD + h) * SS * DHEAD;
  const unsigned char* mrow = mask + (b * SS + q0) * SS;
  for (int kt = w; kt < 32; kt += 4) {
    if (kt + 4 < 32) __builtin_prefetch(Kbase + (kt + 4) * 16 * DHEAD, 0, 2);
    v8f acc = {};
#pragma unroll
    for (int kc = 0; kc < 4; ++kc) {                 // K = 4 x 32 = 128
      v16h a  = frag_a(&Qt[0][0] + kc * 32, DHEAD, lane);
      v16h bf = frag_b(Kbase + (kt * 16) * DHEAD + kc * 32, DHEAD, lane);
      acc = wmma_f16(a, bf, acc);
    }
    int rbase = (lane >> 4) << 3;
    int col = (kt << 4) + (lane & 15);
#pragma unroll
    for (int r = 0; r < 8; ++r) {
      int rr = rbase + r;
      float v = acc[r] * SCALE;
      if (mrow[rr * SS + col]) v = NEGV;
      sc[rr][col] = v;
    }
  }
  __syncthreads();

  // ---- softmax (f32, two-stage reductions; 8 threads per row) ----
  int row = tid >> 3, part = tid & 7, c0 = part << 6;
  float lmax = -3.4e38f;
  for (int c = 0; c < 64; ++c) lmax = fmaxf(lmax, sc[row][c0 + c]);
  pmax[row][part] = lmax;
  __syncthreads();
  float rmax = pmax[row][0];
#pragma unroll
  for (int i = 1; i < 8; ++i) rmax = fmaxf(rmax, pmax[row][i]);
  float lsum = 0.f;
  for (int c = 0; c < 64; ++c) {
    float e = __expf(sc[row][c0 + c] - rmax);
    sc[row][c0 + c] = e;
    lsum += e;
  }
  psum[row][part] = lsum;
  __syncthreads();
  float rsum = 0.f;
#pragma unroll
  for (int i = 0; i < 8; ++i) rsum += psum[row][i];
  float inv = 1.f / rsum;
  float* aout = attn_out + (((b * NHEAD + h) * SS) + q0 + row) * SS + c0;
#pragma unroll
  for (int c = 0; c < 64; ++c) {
    float p = sc[row][c0 + c] * inv;
    aout[c] = p;                                    // materialize attn (f32 output)
    P16[row][c0 + c] = (_Float16)p;                 // f16 copy for WMMA
  }
  __syncthreads();

  // ---- context: C = P (16x512) x V (512x64); wave w owns 16 d-cols ----
  const _Float16* Vbase = Vt + ((b * NHEAD + h) * DHEAD + w * 16) * SS;
  v8f acc = {};
  for (int kc = 0; kc < 16; ++kc) {
    if (kc + 1 < 16) __builtin_prefetch(Vbase + (kc + 1) * 32, 0, 2);
    v16h a  = frag_a(&P16[0][0] + kc * 32, SS, lane);
    v16h bf = frag_b(Vbase + kc * 32, SS, lane);
    acc = wmma_f16(a, bf, acc);
  }
  int rbase = (lane >> 4) << 3, ccol = lane & 15;
#pragma unroll
  for (int r = 0; r < 8; ++r) {
    int s = q0 + rbase + r;
    ctx[(b * SS + s) * HD + h * DHEAD + w * 16 + ccol] = (_Float16)acc[r];
  }
}

// ---------------- Kernel 3: output projection + residual + layernorm ----------
__global__ __launch_bounds__(256) void out_proj_ln_kernel(
    const _Float16* __restrict__ ctx, const _Float16* __restrict__ Wfct,
    const float* __restrict__ inQ, float* __restrict__ out) {
  __shared__ __align__(16) _Float16 At[16][HD];       // 16 KB (ctx tile, contiguous)
  __shared__ __align__(16) float ot[16][DMODEL];      // 8 KB
  __shared__ float pr[16][16][2];
  int bid = blockIdx.x;                 // NB * (SS/16) = 1024
  int b = bid >> 5;
  int s0 = (bid & 31) << 4;
  int tid = threadIdx.x, lane = tid & 31, w = tid >> 5;

  // stage A tile with async global->LDS b128 copies (16 KB = 1024 x 16B)
  const _Float16* Abase = ctx + (b * SS + s0) * HD;
  unsigned lbase = lds_addr32(&At[0][0]);
#pragma unroll
  for (int i = 0; i < 4; ++i) {
    int idx = tid + i * 256;
    async_copy_b128(lbase + idx * 16, (const char*)Abase + idx * 16);
  }
  wait_async0();
  __syncthreads();

  const _Float16* Bbase = Wfct + (w * 16) * HD;
  v8f acc = {};
  for (int kc = 0; kc < 16; ++kc) {                   // K = 512
    v16h a  = frag_a(&At[0][0] + kc * 32, HD, lane);
    v16h bf = frag_b(Bbase + kc * 32, HD, lane);
    acc = wmma_f16(a, bf, acc);
  }
  int rbase = (lane >> 4) << 3, ccol = lane & 15;
#pragma unroll
  for (int r = 0; r < 8; ++r) ot[rbase + r][w * 16 + ccol] = acc[r];
  __syncthreads();

  // residual + layernorm: 16 threads per row, 8 cols each
  int row = tid >> 4, p = tid & 15;
  const float* res = inQ + (b * SS + s0 + row) * DMODEL + p * 8;
  float v[8]; float s = 0.f;
#pragma unroll
  for (int i = 0; i < 8; ++i) { v[i] = ot[row][p * 8 + i] + res[i]; s += v[i]; }
  pr[row][p][0] = s;
  __syncthreads();
  float mean = 0.f;
#pragma unroll
  for (int i = 0; i < 16; ++i) mean += pr[row][i][0];
  mean *= (1.f / 128.f);
  float ss = 0.f;
#pragma unroll
  for (int i = 0; i < 8; ++i) { float d = v[i] - mean; ss += d * d; }
  pr[row][p][1] = ss;
  __syncthreads();
  float var = 0.f;
#pragma unroll
  for (int i = 0; i < 16; ++i) var += pr[row][i][1];
  var *= (1.f / 128.f);
  float rinv = rsqrtf(var + EPSV);
  float* o = out + (b * SS + s0 + row) * DMODEL + p * 8;
#pragma unroll
  for (int i = 0; i < 8; ++i) o[i] = (v[i] - mean) * rinv;
}

// ---------------- host-side launcher ------------------------------------------
extern "C" void kernel_launch(void* const* d_in, const int* in_sizes, int n_in,
                              void* d_out, int out_size, void* d_ws, size_t ws_size,
                              hipStream_t stream) {
  (void)in_sizes; (void)n_in; (void)out_size; (void)ws_size;
  const float* inQ  = (const float*)d_in[0];
  const float* inK  = (const float*)d_in[1];
  const float* inV  = (const float*)d_in[2];
  const unsigned char* mask = (const unsigned char*)d_in[3];
  const float* Wq   = (const float*)d_in[4];
  const float* Wk   = (const float*)d_in[5];
  const float* Wv   = (const float*)d_in[6];
  const float* Wfc  = (const float*)d_in[7];

  float* out  = (float*)d_out;
  float* attn = out + (size_t)NB * SS * DMODEL;

  char* ws = (char*)d_ws;
  const size_t headTensor = (size_t)NB * NHEAD * SS * DHEAD * sizeof(_Float16); // 16 MB
  _Float16* Qh   = (_Float16*)(ws);                       // [b][h][s][d]
  _Float16* Kh   = (_Float16*)(ws + headTensor);          // [b][h][s][d]
  _Float16* Vt   = (_Float16*)(ws + 2 * headTensor);      // [b][h][d][s]
  _Float16* ctx  = (_Float16*)(ws + 3 * headTensor);      // [b][s][h*64+d]
  _Float16* Wtq  = (_Float16*)(ws + 4 * headTensor);
  _Float16* Wtk  = Wtq + 512 * 128;
  _Float16* Wtv  = Wtk + 512 * 128;
  _Float16* Wfct = Wtv + 512 * 128;

  prep_weights_kernel<<<256, 256, 0, stream>>>(Wq, Wk, Wv, Wfc, Wtq, Wtk, Wtv, Wfct);
  qkv_proj_kernel<<<NB * (SS / 16), 256, 0, stream>>>(inQ, inK, inV, Wtq, Wtk, Wtv,
                                                      Qh, Kh, Vt);
  attn_kernel<<<NB * NHEAD * (SS / 16), 128, 0, stream>>>(Qh, Kh, Vt, mask, attn, ctx);
  out_proj_ln_kernel<<<NB * (SS / 16), 256, 0, stream>>>(ctx, Wfct, inQ, out);
}
